// ScannedBiRNN_74895639708014
// MI455X (gfx1250) — matrix operand order repaired and checked by reading, not compiled, not run on hardware
//
#include <hip/hip_runtime.h>
#include <hip/hip_bf16.h>

// ScannedBiRNN for MI455X (gfx1250, wave32, WMMA + async-to-LDS).
// Phase 1: x_proj = feats @ Wx + bx (both dirs) via v_wmma_f32_16x16x32_bf16 -> bf16 scratch.
// Phase 2: persistent cooperative recurrence, 8 WGs/direction, Wh slice resident in LDS (bf16),
//          per-step h broadcast via GLOBAL_LOAD_ASYNC_TO_LDS_B128 of a bf16 h mirror,
//          per-step device-scope barrier via monotonic atomic counter, double-buffered h.

#define T_STEPS 512
#define BATCH   64
#define DIN     256
#define HID     256
#define H3      768
#define TBROWS  (T_STEPS * BATCH)   // 32768
#define NWG     8                   // workgroups per direction in recurrence

typedef __bf16 v16bf __attribute__((ext_vector_type(16)));
typedef float  v8f   __attribute__((ext_vector_type(8)));

union FragB { v16bf v; unsigned int u[8]; };
union FragC { v8f  v; float f[8]; };

__device__ __forceinline__ unsigned short f2bf(float x) {
    unsigned int u = __float_as_uint(x);
    u += 0x7FFFu + ((u >> 16) & 1u);           // round-to-nearest-even
    return (unsigned short)(u >> 16);
}
__device__ __forceinline__ unsigned int pack2(float a, float b) {
    return (unsigned int)f2bf(a) | ((unsigned int)f2bf(b) << 16);
}
__device__ __forceinline__ float bf2f(unsigned short h) {
    return __uint_as_float(((unsigned int)h) << 16);
}
__device__ __forceinline__ float sigmoidf_(float x) { return 1.0f / (1.0f + __expf(-x)); }

// Async copy 16B/lane from global (saddr + voff) into LDS at lds_addr. ASYNCcnt-tracked.
__device__ __forceinline__ void async_load_b128(unsigned lds_addr, unsigned voff, const void* saddr) {
    asm volatile("global_load_async_to_lds_b128 %0, %1, %2"
                 :: "v"(lds_addr), "v"(voff), "s"(saddr) : "memory");
}
__device__ __forceinline__ void wait_async0() {
    asm volatile("s_wait_asynccnt 0x0" ::: "memory");
}

// ---------------------------------------------------------------------------
// init: zero the two per-direction barrier counters (ws is NOT re-poisoned
// between graph replays, so this must run every launch).
// ---------------------------------------------------------------------------
__global__ void birnn_init_kernel(unsigned int* counters) {
    if (threadIdx.x < 2) counters[threadIdx.x] = 0u;
}

// ---------------------------------------------------------------------------
// Phase 1: x_proj GEMM. Grid: (TBROWS/64, H3/64, 2 dirs). Block: 256 (8 waves).
// ---------------------------------------------------------------------------
__global__ __launch_bounds__(256) void birnn_xproj_kernel(
    const float* __restrict__ feats,
    const float* __restrict__ Wx_f, const float* __restrict__ bx_f,
    const float* __restrict__ Wx_b, const float* __restrict__ bx_b,
    unsigned short* __restrict__ xp /* [2][TBROWS][768] bf16 */)
{
    const int dir = blockIdx.z;
    const float* Wx = dir ? Wx_b : Wx_f;
    const float* bx = dir ? bx_b : bx_f;
    unsigned short* out = xp + (size_t)dir * (size_t)TBROWS * H3;

    const int row0 = blockIdx.x * 64;
    const int c0   = blockIdx.y * 64;

    __shared__ unsigned int ldsA[64][128];  // [m][kpair] packed bf16 pairs, 32 KB
    __shared__ unsigned int ldsW[64][128];  // [n][kpair] column-major-packed, 32 KB

    const int tid = threadIdx.x;

    for (int i = tid; i < 64 * 128; i += 256) {
        const int kp = i & 127, m = i >> 7;
        const float* p = feats + (size_t)(row0 + m) * DIN + kp * 2;
        ldsA[m][kp] = pack2(p[0], p[1]);
    }
    for (int i = tid; i < 64 * 128; i += 256) {
        const int n = i & 63, kp = i >> 6;
        const float w0 = Wx[(size_t)(2 * kp)     * H3 + c0 + n];
        const float w1 = Wx[(size_t)(2 * kp + 1) * H3 + c0 + n];
        ldsW[n][kp] = pack2(w0, w1);
    }
    __syncthreads();

    const int wave = tid >> 5, lane = tid & 31;
    const int mt = wave >> 1, g = wave & 1;     // 4 M-tiles x 2 N-groups
    const int lm = lane & 15, hi = lane >> 4;

    FragC acc0, acc1;
#pragma unroll
    for (int v = 0; v < 8; ++v) { acc0.f[v] = 0.0f; acc1.f[v] = 0.0f; }

#pragma unroll
    for (int kt = 0; kt < 8; ++kt) {
        FragB a, b0, b1;
#pragma unroll
        for (int v = 0; v < 8; ++v) {
            const int kpl = (v < 4) ? (v + hi * 4) : (8 + (v - 4) + hi * 4);
            a.u[v] = ldsA[mt * 16 + lm][kt * 16 + kpl];
        }
#pragma unroll
        for (int v = 0; v < 8; ++v) {
            const int kpb = (v < 4) ? (v + hi * 4) : (8 + (v - 4) + hi * 4);
            b0.u[v] = ldsW[(g * 2 + 0) * 16 + lm][kt * 16 + kpb];
            b1.u[v] = ldsW[(g * 2 + 1) * 16 + lm][kt * 16 + kpb];
        }
        acc0.v = __builtin_amdgcn_wmma_f32_16x16x32_bf16(false, a.v, false, b0.v,
                                                         (short)0, acc0.v, false, false);
        acc1.v = __builtin_amdgcn_wmma_f32_16x16x32_bf16(false, a.v, false, b1.v,
                                                         (short)0, acc1.v, false, false);
    }

#pragma unroll
    for (int half = 0; half < 2; ++half) {
        FragC& acc = half ? acc1 : acc0;
        const int nt  = g * 2 + half;
        const int col = c0 + nt * 16 + lm;
        const float bias = bx[col];
#pragma unroll
        for (int v = 0; v < 8; ++v) {
            const int m = mt * 16 + v + hi * 8;
            out[(size_t)(row0 + m) * H3 + col] = f2bf(acc.f[v] + bias);
        }
    }
}

// ---------------------------------------------------------------------------
// Phase 2: cooperative recurrence. Grid: 2*NWG blocks of 256 threads, all
// co-resident. Block (dir, wg) owns hidden columns [wg*32, wg*32+32).
// ---------------------------------------------------------------------------
__global__ __launch_bounds__(256) void birnn_recur_kernel(
    const int*   __restrict__ resets,   // [T][B]
    const float* __restrict__ h_fwd0,   // [B][H]
    const float* __restrict__ h_bwd0,   // [B][H]
    const float* __restrict__ Wh_f,     // [H][3H]
    const float* __restrict__ bhn_f,    // [H]
    const float* __restrict__ Wh_b,
    const float* __restrict__ bhn_b,
    const unsigned short* __restrict__ xp,  // [2][TBROWS][768] bf16
    float* __restrict__ hbuf,               // [2 dirs][2 phases][B*H] fp32
    unsigned short* __restrict__ hbfbuf,    // [2 dirs][2 phases][B*H] bf16 mirror
    unsigned int* __restrict__ counters,
    float* __restrict__ d_out)          // hf[B*H] | hb[B*H] | embs[T][B][2H]
{
    const int dir = blockIdx.x / NWG;
    const int wg  = blockIdx.x % NWG;

    const float* Wh   = dir ? Wh_b  : Wh_f;
    const float* bhnp = dir ? bhn_b : bhn_f;
    const float* h0   = dir ? h_bwd0 : h_fwd0;
    const unsigned short* xpd = xp + (size_t)dir * (size_t)TBROWS * H3;
    float*          hb   = hbuf   + (size_t)dir * 2 * (BATCH * HID);
    unsigned short* hbfd = hbfbuf + (size_t)dir * 2 * (BATCH * HID);
    unsigned int* ctr = counters + dir;
    float* out_h    = d_out + (size_t)dir * (BATCH * HID);
    float* out_embs = d_out + (size_t)2 * BATCH * HID;

    const int j0  = wg * 32;
    const int tid = threadIdx.x;

    __shared__ unsigned int lds_h[64][128];  // h bf16 packed pairs, 32 KB
    __shared__ unsigned int lds_w[96][128];  // Wh slice col-major packed, 48 KB

    // stage the Wh slice once (loop-invariant)
    for (int i = tid; i < 96 * 128; i += 256) {
        const int n = i % 96, kp = i / 96;
        int gcol;
        if (n < 32)      gcol =            j0 + n;         // r
        else if (n < 64) gcol = HID      + j0 + (n - 32);  // z
        else             gcol = 2 * HID  + j0 + (n - 64);  // n
        const float w0 = Wh[(size_t)(2 * kp)     * H3 + gcol];
        const float w1 = Wh[(size_t)(2 * kp + 1) * H3 + gcol];
        lds_w[n][kp] = pack2(w0, w1);
    }

    const int wave = tid >> 5, lane = tid & 31;
    const int mt = wave >> 1, g = wave & 1;
    const int lm = lane & 15, hi = lane >> 4;
    const int j  = j0 + g * 16 + lm;            // this lane's hidden column
    const float bn = bhnp[j];
    const unsigned ldsbase = (unsigned)(size_t)(&lds_h[0][0]);

    for (int s = 0; s < T_STEPS; ++s) {
        const int t_idx = dir ? (T_STEPS - 1 - s) : s;
        const float* hsrc = (s == 0) ? h0 : (hb + (size_t)(s & 1) * (BATCH * HID));
        float*          hdst  = hb   + (size_t)((s + 1) & 1) * (BATCH * HID);
        const unsigned short* hbsrc = hbfd + (size_t)(s & 1) * (BATCH * HID);
        unsigned short*       hbdst = hbfd + (size_t)((s + 1) & 1) * (BATCH * HID);

        // phase A: h (bf16) -> LDS, reset-masked
        if (s == 0) {
            // h0 arrives fp32: convert + mask on the fly
            for (int i = tid; i < 64 * 128; i += 256) {
                const int kp = i & 127, m = i >> 7;
                float a0 = 0.0f, a1 = 0.0f;
                if (!resets[t_idx * BATCH + m]) {
                    a0 = h0[m * HID + 2 * kp];
                    a1 = h0[m * HID + 2 * kp + 1];
                }
                lds_h[m][kp] = pack2(a0, a1);
            }
            __syncthreads();
        } else {
            // async DMA of the bf16 h mirror straight into LDS (bypasses VGPRs)
#pragma unroll
            for (int it = 0; it < 8; ++it) {
                const unsigned voff = (unsigned)(tid + it * 256) * 16u;
                async_load_b128(ldsbase + voff, voff, (const void*)hbsrc);
            }
            wait_async0();
            __syncthreads();
            // sparse reset masking (~10% of rows): zero masked batch rows in LDS
            for (int i = tid; i < 64 * 128; i += 256) {
                const int kp = i & 127, m = i >> 7;
                if (resets[t_idx * BATCH + m]) lds_h[m][kp] = 0u;
            }
            __syncthreads();
        }

        // phase B: P = h @ Wh_slice  (3 gate tiles per wave)
        FragC ar, az, an_;
#pragma unroll
        for (int v = 0; v < 8; ++v) { ar.f[v] = 0.0f; az.f[v] = 0.0f; an_.f[v] = 0.0f; }

#pragma unroll
        for (int kt = 0; kt < 8; ++kt) {
            FragB a, br, bz, bnn;
#pragma unroll
            for (int v = 0; v < 8; ++v) {
                const int kpl = (v < 4) ? (v + hi * 4) : (8 + (v - 4) + hi * 4);
                a.u[v] = lds_h[mt * 16 + lm][kt * 16 + kpl];
            }
#pragma unroll
            for (int v = 0; v < 8; ++v) {
                const int kpb = kt * 16 + ((v < 4) ? (v + hi * 4) : (8 + (v - 4) + hi * 4));
                br.u[v]  = lds_w[     g * 16 + lm][kpb];
                bz.u[v]  = lds_w[32 + g * 16 + lm][kpb];
                bnn.u[v] = lds_w[64 + g * 16 + lm][kpb];
            }
            ar.v  = __builtin_amdgcn_wmma_f32_16x16x32_bf16(false, a.v, false, br.v,
                                                            (short)0, ar.v,  false, false);
            az.v  = __builtin_amdgcn_wmma_f32_16x16x32_bf16(false, a.v, false, bz.v,
                                                            (short)0, az.v,  false, false);
            an_.v = __builtin_amdgcn_wmma_f32_16x16x32_bf16(false, a.v, false, bnn.v,
                                                            (short)0, an_.v, false, false);
        }

        // phase C: gate math + stores (fp32 h, bf16 h mirror, embs)
        const unsigned short* xrow0 = xpd + (size_t)t_idx * BATCH * H3;
#pragma unroll
        for (int v = 0; v < 8; ++v) {
            const int m = mt * 16 + v + hi * 8;
            const unsigned short* xr_p = xrow0 + (size_t)m * H3;
            const float xr = bf2f(xr_p[j]);
            const float xz = bf2f(xr_p[HID + j]);
            const float xn = bf2f(xr_p[2 * HID + j]);
            const float r  = sigmoidf_(xr + ar.f[v]);
            const float z  = sigmoidf_(xz + az.f[v]);
            const float hp = resets[t_idx * BATCH + m] ? 0.0f : hsrc[m * HID + j];
            const float nn = tanhf(xn + r * (an_.f[v] + bn));
            const float hnew = (1.0f - z) * nn + z * hp;
            hdst[m * HID + j]  = hnew;
            hbdst[m * HID + j] = f2bf(hnew);
            out_embs[(size_t)s * BATCH * (2 * HID) + (size_t)m * (2 * HID) + dir * HID + j] = hnew;
            if (s == T_STEPS - 1) out_h[m * HID + j] = hnew;
        }

        // phase D: device-scope step barrier (monotonic counter)
        __threadfence();
        __syncthreads();
        if (tid == 0) {
            __hip_atomic_fetch_add(ctr, 1u, __ATOMIC_RELEASE, __HIP_MEMORY_SCOPE_AGENT);
            const unsigned int target = (unsigned int)(NWG * (s + 1));
            while (__hip_atomic_load(ctr, __ATOMIC_ACQUIRE, __HIP_MEMORY_SCOPE_AGENT) < target) {
                __builtin_amdgcn_s_sleep(1);
            }
        }
        __syncthreads();
    }
}

// ---------------------------------------------------------------------------
extern "C" void kernel_launch(void* const* d_in, const int* in_sizes, int n_in,
                              void* d_out, int out_size, void* d_ws, size_t ws_size,
                              hipStream_t stream) {
    const float* feats  = (const float*)d_in[0];
    const int*   resets = (const int*)  d_in[1];
    const float* h_fwd0 = (const float*)d_in[2];
    const float* h_bwd0 = (const float*)d_in[3];
    const float* Wx_f   = (const float*)d_in[4];
    const float* Wh_f   = (const float*)d_in[5];
    const float* bx_f   = (const float*)d_in[6];
    const float* bhn_f  = (const float*)d_in[7];
    const float* Wx_b   = (const float*)d_in[8];
    const float* Wh_b   = (const float*)d_in[9];
    const float* bx_b   = (const float*)d_in[10];
    const float* bhn_b  = (const float*)d_in[11];

    // ws layout:
    //   [0, 256)                      barrier counters
    //   [256, 256+256KB)              fp32 h double buffers (2 dirs x 2 phases)
    //   [+0, +128KB)                  bf16 h mirrors        (2 dirs x 2 phases)
    //   [..., +~96MB)                 bf16 x_proj scratch 2 x TBROWS x 768
    char* ws = (char*)d_ws;
    unsigned int*   counters = (unsigned int*)ws;
    float*          hbuf     = (float*)(ws + 256);
    unsigned short* hbf      = (unsigned short*)(ws + 256 + (size_t)2 * 2 * BATCH * HID * 4);
    unsigned short* xp       = (unsigned short*)(ws + 256 + (size_t)2 * 2 * BATCH * HID * 4
                                                          + (size_t)2 * 2 * BATCH * HID * 2);
    float* out = (float*)d_out;

    hipLaunchKernelGGL(birnn_init_kernel, dim3(1), dim3(32), 0, stream, counters);
    hipLaunchKernelGGL(birnn_xproj_kernel, dim3(TBROWS / 64, H3 / 64, 2), dim3(256), 0, stream,
                       feats, Wx_f, bx_f, Wx_b, bx_b, xp);
    hipLaunchKernelGGL(birnn_recur_kernel, dim3(2 * NWG), dim3(256), 0, stream,
                       resets, h_fwd0, h_bwd0, Wh_f, bhn_f, Wh_b, bhn_b,
                       xp, hbuf, hbf, counters, out);
}
